// SequenceFrameTransformerUpdate_59115929862892
// MI455X (gfx1250) — matrix operand order repaired and checked by reading, not compiled, not run on hardware
//
#include <hip/hip_runtime.h>
#include <stdint.h>

typedef __bf16 bf16_t;
typedef __attribute__((ext_vector_type(16))) __bf16 v16bf;
typedef __attribute__((ext_vector_type(8)))  float  v8f;

#define BB 2
#define NN 1024
#define CS 256
#define HH 8
#define DHD 32
#define CZ 64
#define NRES 256
#define CCOND 384
#define MM (BB*NN)          // 2048
#define INFV 100000.0f

__device__ __forceinline__ float warp_sum(float v) {
#pragma unroll
  for (int off = 16; off > 0; off >>= 1) v += __shfl_xor(v, off, 32);
  return v;
}
__device__ __forceinline__ float warp_max(float v) {
#pragma unroll
  for (int off = 16; off > 0; off >>= 1) v = fmaxf(v, __shfl_xor(v, off, 32));
  return v;
}
__device__ __forceinline__ float sigmoidf(float x) { return 1.0f / (1.0f + __expf(-x)); }

// ---- weight transpose: W (K x N) f32 -> Wt (N x K) bf16 ----
__global__ void k_transpose_bf16(const float* __restrict__ W, bf16_t* __restrict__ Wt,
                                 int K, int N) {
  int i = blockIdx.x * blockDim.x + threadIdx.x;
  if (i >= K * N) return;
  int n = i / K, k = i - n * K;
  Wt[i] = (bf16_t)W[(size_t)k * N + n];
}

__global__ void k_convert_bf16(const float* __restrict__ X, bf16_t* __restrict__ Y, int n) {
  int i = blockIdx.x * blockDim.x + threadIdx.x;
  if (i < n) Y[i] = (bf16_t)X[i];
}

// ---- layernorm, one wave32 per row, g/b optional ----
__global__ void k_layernorm(const float* __restrict__ X, const float* __restrict__ g,
                            const float* __restrict__ bset, bf16_t* __restrict__ Y,
                            int rows, int C) {
  int wid = (blockIdx.x * blockDim.x + threadIdx.x) >> 5;
  if (wid >= rows) return;
  int lane = threadIdx.x & 31;
  const float* x = X + (size_t)wid * C;
  float s = 0.f, ss = 0.f;
  for (int c = lane; c < C; c += 32) { float v = x[c]; s += v; ss += v * v; }
  s = warp_sum(s); ss = warp_sum(ss);
  float mu = s / (float)C;
  float var = ss / (float)C - mu * mu;
  float rs = rsqrtf(var + 1e-5f);
  bf16_t* y = Y + (size_t)wid * C;
  for (int c = lane; c < C; c += 32) {
    float v = (x[c] - mu) * rs;
    if (g)    v *= g[c];
    if (bset) v += bset[c];
    y[c] = (bf16_t)v;
  }
}

// ---- generic bf16 WMMA GEMM: C(MxN,f32) = A(MxK,bf16) * Bt(NxK,bf16)^T + bias ----
// one wave32 per 16x16 tile; K multiple of 32
__global__ void k_gemm_bf16(const bf16_t* __restrict__ A, const bf16_t* __restrict__ Bt,
                            const float* __restrict__ bias, float* __restrict__ C,
                            int M, int N, int K, int lda, int ldb, int ldc) {
  int wid = (blockIdx.x * blockDim.x + threadIdx.x) >> 5;
  int tiles_n = N >> 4;
  int total = (M >> 4) * tiles_n;
  if (wid >= total) return;
  int tm = wid / tiles_n, tn = wid - tm * tiles_n;
  int lane = threadIdx.x & 31;
  int lh = lane >> 4, lm = lane & 15;

  union { v16bf v; uint32_t u[8]; } a, b;
  v8f acc = {0.f,0.f,0.f,0.f,0.f,0.f,0.f,0.f};
  const bf16_t* arow = A  + (size_t)(tm * 16 + lm) * lda;
  const bf16_t* brow = Bt + (size_t)(tn * 16 + lm) * ldb + lh * 16;
  for (int k0 = 0; k0 < K; k0 += 32) {
#pragma unroll
    for (int v = 0; v < 8; ++v) {
      int ka = (lh << 4) + (v << 1);   // A: K pairs per VGPR
      a.u[v] = *reinterpret_cast<const uint32_t*>(arow + k0 + ka);
      b.u[v] = *reinterpret_cast<const uint32_t*>(brow + k0 + 2 * v); // B: K pairs per VGPR
    }
    acc = __builtin_amdgcn_wmma_f32_16x16x32_bf16(false, a.v, false, b.v,
                                                  (short)0, acc, false, false);
  }
  int n = tn * 16 + lm;
  float bv = bias ? bias[n] : 0.0f;
#pragma unroll
  for (int r = 0; r < 8; ++r) {
    int m = tm * 16 + lh * 8 + r;
    C[(size_t)m * ldc + n] = acc[r] + bv;
  }
}

// ---- q/kv split + bf16, and transposed V: vt[b][h][d][j] ----
__global__ void k_prep_attn(const float* __restrict__ q, const float* __restrict__ kv,
                            bf16_t* __restrict__ qb, bf16_t* __restrict__ kb,
                            bf16_t* __restrict__ vt) {
  int i = blockIdx.x * blockDim.x + threadIdx.x;
  if (i >= MM * CS) return;
  int m = i >> 8, c = i & 255;
  qb[i] = (bf16_t)q[i];
  kb[i] = (bf16_t)kv[(size_t)m * 512 + c];
  float vv = kv[(size_t)m * 512 + 256 + c];
  int b = m >> 10, j = m & 1023, h = c >> 5, d = c & 31;
  vt[(((size_t)(b * HH + h) * DHD + d) << 10) + j] = (bf16_t)vv;
}

// ---- pair bias via WMMA ----
// block = 256 thr = 8 waves; each wave: 16 rows (fixed b,i; 16 consecutive j).
// LN stats by lane pairs (2 shuffles/row), normalized bf16 tile -> LDS,
// 2x v_wmma_f32_16x16x32_bf16 against zero-padded Wbias^T tile in LDS.
// att[b][h][i][j] = LN(fp[b,i,j,:])@Wbias[:,h] + (m_i*m_j-1)*INF
__global__ void k_pbias(const float* __restrict__ fp, const float* __restrict__ lg,
                        const float* __restrict__ lb, const float* __restrict__ Wbias,
                        const float* __restrict__ mask, float* __restrict__ att) {
  __shared__ bf16_t sWbT[16 * 64];        // WbT[n][k], n>=8 zero-padded
  __shared__ float  sg[CZ], sb[CZ];
  __shared__ bf16_t tile[8][16 * 64];     // per-wave normalized 16x64 tiles
  int t = threadIdx.x;
  if (t < CZ) { sg[t] = lg[t]; sb[t] = lb[t]; }
  for (int e = t; e < 16 * 64; e += 256) {
    int n = e >> 6, k = e & 63;
    sWbT[e] = (bf16_t)((n < 8) ? Wbias[k * 8 + n] : 0.0f);
  }
  __syncthreads();

  int wave = t >> 5, lane = t & 31;
  int lh = lane >> 4, lm = lane & 15;
  long long rowbase = (long long)blockIdx.x * 128 + wave * 16; // 128-row blocks share (b,i)
  int b  = (int)(rowbase >> 20);
  int ij = (int)(rowbase & ((1 << 20) - 1));
  int i = ij >> 10, j0 = ij & 1023;

  // lane pair (lm, lm+16) handles row lm; each covers 32 of 64 elems
  const float* rp = fp + (rowbase + lm) * CZ + lh * 32;
  float vals[32];
  float s = 0.f, ss = 0.f;
#pragma unroll
  for (int e = 0; e < 32; e += 4) {
    float4 q4 = *reinterpret_cast<const float4*>(rp + e);
    vals[e] = q4.x; vals[e+1] = q4.y; vals[e+2] = q4.z; vals[e+3] = q4.w;
    s  += q4.x + q4.y + q4.z + q4.w;
    ss += q4.x*q4.x + q4.y*q4.y + q4.z*q4.z + q4.w*q4.w;
  }
  s  += __shfl_xor(s, 16, 32);
  ss += __shfl_xor(ss, 16, 32);
  float mu = s * (1.0f / CZ);
  float rs = rsqrtf(ss * (1.0f / CZ) - mu * mu + 1e-5f);

  bf16_t* trow = &tile[wave][lm * 64 + lh * 32];
#pragma unroll
  for (int e = 0; e < 32; e += 2) {
    int c = lh * 32 + e;
    union { bf16_t h[2]; uint32_t u; } p;
    p.h[0] = (bf16_t)((vals[e]   - mu) * rs * sg[c]     + sb[c]);
    p.h[1] = (bf16_t)((vals[e+1] - mu) * rs * sg[c + 1] + sb[c + 1]);
    *reinterpret_cast<uint32_t*>(trow + e) = p.u;
  }
  // same-wave LDS producer/consumer: DS ops are in-order per wave

  union { v16bf v; uint4 q[2]; } a, bq;
  v8f acc = {0.f,0.f,0.f,0.f,0.f,0.f,0.f,0.f};
#pragma unroll
  for (int kb = 0; kb < CZ; kb += 32) {
    const bf16_t* ar = &tile[wave][lm * 64 + kb + lh * 16];
    a.q[0] = *reinterpret_cast<const uint4*>(ar);        // k = kb+lh*16  .. +7
    a.q[1] = *reinterpret_cast<const uint4*>(ar + 8);    // k = kb+lh*16+8.. +15
    const bf16_t* br = &sWbT[lm * 64 + kb + lh * 16];
    bq.q[0] = *reinterpret_cast<const uint4*>(br);       // v0..v3
    bq.q[1] = *reinterpret_cast<const uint4*>(br + 8);   // v4..v7
    acc = __builtin_amdgcn_wmma_f32_16x16x32_bf16(false, a.v, false, bq.v,
                                                  (short)0, acc, false, false);
  }

  // C: VGPR r -> local row (j offset) = lh*8+r, col = head (lm<8 valid)
  float mi = mask[b * NN + i];
  if (lm < 8) {
    int h = lm;
    size_t obase = (((size_t)(b * HH + h)) * NN + i) * NN + j0 + lh * 8;
#pragma unroll
    for (int r = 0; r < 8; ++r) {
      int j = j0 + lh * 8 + r;
      float mterm = (mi * mask[b * NN + j] - 1.0f) * INFV;
      att[obase + r] = acc[r] + mterm;
    }
  }
}

// ---- attention logits: att += (q k^T)/sqrt(DH); one WMMA (K=32) per 16x16 tile ----
__global__ void k_attn_logits(const bf16_t* __restrict__ qb, const bf16_t* __restrict__ kb,
                              float* __restrict__ att) {
  int wid = (blockIdx.x * blockDim.x + threadIdx.x) >> 5;
  if (wid >= BB * HH * 64 * 64) return;
  int bh = wid >> 12;
  int tt = wid & 4095;
  int ti = tt >> 6, tj = tt & 63;
  int b = bh >> 3, h = bh & 7;
  int lane = threadIdx.x & 31;
  int lh = lane >> 4, lm = lane & 15;

  union { v16bf v; uint32_t u[8]; } a, bq;
  const bf16_t* arow = qb + (size_t)(b * NN + ti * 16 + lm) * CS + h * DHD;
  const bf16_t* brow = kb + (size_t)(b * NN + tj * 16 + lm) * CS + h * DHD + lh * 16;
#pragma unroll
  for (int v = 0; v < 8; ++v) {
    int ka = (lh << 4) + (v << 1);
    a.u[v]  = *reinterpret_cast<const uint32_t*>(arow + ka);
    bq.u[v] = *reinterpret_cast<const uint32_t*>(brow + 2 * v);
  }
  v8f acc = {0.f,0.f,0.f,0.f,0.f,0.f,0.f,0.f};
  acc = __builtin_amdgcn_wmma_f32_16x16x32_bf16(false, a.v, false, bq.v,
                                                (short)0, acc, false, false);
  const float scale = 0.17677669529663688f; // 1/sqrt(32)
#pragma unroll
  for (int r = 0; r < 8; ++r) {
    size_t off = (((size_t)bh * NN) + ti * 16 + lh * 8 + r) * NN + tj * 16 + lm;
    att[off] += acc[r] * scale;
  }
}

// ---- softmax over rows of 1024, block(256) per row, bf16 output ----
__global__ void k_softmax(const float* __restrict__ att, bf16_t* __restrict__ attb) {
  __shared__ float red[8];
  size_t row = blockIdx.x;
  const float* x = att + row * NN;
  int t = threadIdx.x;
  float v0 = x[t], v1 = x[t + 256], v2 = x[t + 512], v3 = x[t + 768];
  float mx = fmaxf(fmaxf(v0, v1), fmaxf(v2, v3));
  mx = warp_max(mx);
  if ((t & 31) == 0) red[t >> 5] = mx;
  __syncthreads();
  float m = fmaxf(fmaxf(fmaxf(red[0], red[1]), fmaxf(red[2], red[3])),
                  fmaxf(fmaxf(red[4], red[5]), fmaxf(red[6], red[7])));
  __syncthreads();
  float e0 = __expf(v0 - m), e1 = __expf(v1 - m), e2 = __expf(v2 - m), e3 = __expf(v3 - m);
  float s = warp_sum(e0 + e1 + e2 + e3);
  if ((t & 31) == 0) red[t >> 5] = s;
  __syncthreads();
  float stot = red[0] + red[1] + red[2] + red[3] + red[4] + red[5] + red[6] + red[7];
  float inv = 1.0f / stot;
  bf16_t* y = attb + row * NN;
  y[t]       = (bf16_t)(e0 * inv);
  y[t + 256] = (bf16_t)(e1 * inv);
  y[t + 512] = (bf16_t)(e2 * inv);
  y[t + 768] = (bf16_t)(e3 * inv);
}

// ---- elementwise kernels ----
__global__ void k_ew_gate(const float* __restrict__ o, const float* __restrict__ gp,
                          bf16_t* __restrict__ u, int n) {
  int i = blockIdx.x * blockDim.x + threadIdx.x;
  if (i < n) u[i] = (bf16_t)(o[i] * sigmoidf(gp[i]));
}
__global__ void k_ew_rig(const float* __restrict__ re, const float* __restrict__ upd,
                         const float* __restrict__ mask, float* __restrict__ rig, int n) {
  int i = blockIdx.x * blockDim.x + threadIdx.x;
  if (i < n) rig[i] = re[i] + upd[i] * mask[i >> 8];
}
__global__ void k_ew_sa(const bf16_t* __restrict__ sn, const float* __restrict__ cond1,
                        const float* __restrict__ condn, const int* __restrict__ idx,
                        const float* __restrict__ mask, bf16_t* __restrict__ sa, int n) {
  int i = blockIdx.x * blockDim.x + threadIdx.x;
  if (i >= n) return;
  int m = i >> 8, c = i & 255;
  int b = m >> 10;
  int id = idx[m];
  float mk = mask[m];
  size_t goff = ((size_t)(b * NRES + id)) * CS + c;
  float g  = cond1[goff] * mk;
  float bs = condn[goff] * mk;
  sa[i] = (bf16_t)((float)sn[i] * sigmoidf(g) + bs);
}
__global__ void k_ew_bmid(const float* __restrict__ b1, const float* __restrict__ b2,
                          bf16_t* __restrict__ bmid, int n) {
  int i = blockIdx.x * blockDim.x + threadIdx.x;
  if (i < n) { float v = b1[i]; bmid[i] = (bf16_t)(v * sigmoidf(v) * b2[i]); }
}
__global__ void k_ew_final(const float* __restrict__ rig, const float* __restrict__ tmid,
                           const float* __restrict__ cgp, const int* __restrict__ idx,
                           const float* __restrict__ mask, float* __restrict__ out, int n) {
  int i = blockIdx.x * blockDim.x + threadIdx.x;
  if (i >= n) return;
  int m = i >> 8, c = i & 255;
  int b = m >> 10;
  int id = idx[m];
  float mk = mask[m];
  float cg = cgp[((size_t)(b * NRES + id)) * CS + c] * mk;
  out[i] = rig[i] + sigmoidf(cg) * tmid[i] * mk;
}

extern "C" void kernel_launch(void* const* d_in, const int* in_sizes, int n_in,
                              void* d_out, int out_size, void* d_ws, size_t ws_size,
                              hipStream_t stream) {
  (void)in_sizes; (void)n_in; (void)out_size; (void)ws_size;
  const float* s_in  = (const float*)d_in[0];
  const float* re    = (const float*)d_in[1];
  const float* fp    = (const float*)d_in[2];
  const float* mask  = (const float*)d_in[3];
  const int*   idx   = (const int*)  d_in[4];
  const float* ln_g  = (const float*)d_in[5];
  const float* ln_b  = (const float*)d_in[6];
  const float* Wq    = (const float*)d_in[7];
  const float* Wkv   = (const float*)d_in[8];
  const float* lnz_g = (const float*)d_in[9];
  const float* lnz_b = (const float*)d_in[10];
  const float* Wbias = (const float*)d_in[11];
  const float* Wg    = (const float*)d_in[12];
  const float* Wo    = (const float*)d_in[13];
  const float* lncg  = (const float*)d_in[14];
  const float* Wc1   = (const float*)d_in[15];
  const float* bc1   = (const float*)d_in[16];
  const float* Wcn   = (const float*)d_in[17];
  const float* W1    = (const float*)d_in[18];
  const float* W2    = (const float*)d_in[19];
  const float* Wc2   = (const float*)d_in[20];
  const float* bc2   = (const float*)d_in[21];
  const float* Wb    = (const float*)d_in[22];
  float* out = (float*)d_out;

  char* ws = (char*)d_ws;
  size_t off = 0;
  auto alloc = [&](size_t bytes) -> void* {
    void* p = ws + off;
    off += (bytes + 255) & ~(size_t)255;
    return p;
  };
  bf16_t* WqT  = (bf16_t*)alloc((size_t)CS * CS * 2);
  bf16_t* WkvT = (bf16_t*)alloc((size_t)2 * CS * CS * 2);
  bf16_t* WgT  = (bf16_t*)alloc((size_t)CS * CS * 2);
  bf16_t* WoT  = (bf16_t*)alloc((size_t)CS * CS * 2);
  bf16_t* Wc1T = (bf16_t*)alloc((size_t)CS * CCOND * 2);
  bf16_t* WcnT = (bf16_t*)alloc((size_t)CS * CCOND * 2);
  bf16_t* Wc2T = (bf16_t*)alloc((size_t)CS * CCOND * 2);
  bf16_t* W1T  = (bf16_t*)alloc((size_t)2 * CS * CS * 2);
  bf16_t* W2T  = (bf16_t*)alloc((size_t)2 * CS * CS * 2);
  bf16_t* WbT  = (bf16_t*)alloc((size_t)CS * 2 * CS * 2);
  bf16_t* xn   = (bf16_t*)alloc((size_t)MM * CS * 2);
  float*  qf   = (float*) alloc((size_t)MM * CS * 4);
  float*  kvf  = (float*) alloc((size_t)MM * 2 * CS * 4);
  float*  gpf  = (float*) alloc((size_t)MM * CS * 4);
  bf16_t* qb   = (bf16_t*)alloc((size_t)MM * CS * 2);
  bf16_t* kb   = (bf16_t*)alloc((size_t)MM * CS * 2);
  bf16_t* vt   = (bf16_t*)alloc((size_t)BB * HH * DHD * NN * 2);
  float*  att  = (float*) alloc((size_t)BB * HH * NN * NN * 4);
  bf16_t* attb = (bf16_t*)alloc((size_t)BB * HH * NN * NN * 2);
  float*  of   = (float*) alloc((size_t)MM * CS * 4);
  bf16_t* ub   = (bf16_t*)alloc((size_t)MM * CS * 2);
  float*  updf = (float*) alloc((size_t)MM * CS * 4);
  float*  rig  = (float*) alloc((size_t)MM * CS * 4);
  bf16_t* snb  = (bf16_t*)alloc((size_t)MM * CS * 2);
  bf16_t* sb   = (bf16_t*)alloc((size_t)BB * NRES * CCOND * 2);
  bf16_t* cnb  = (bf16_t*)alloc((size_t)BB * NRES * CCOND * 2);
  float*  c1f  = (float*) alloc((size_t)BB * NRES * CS * 4);
  float*  cnf  = (float*) alloc((size_t)BB * NRES * CS * 4);
  float*  cgpf = (float*) alloc((size_t)BB * NRES * CS * 4);
  bf16_t* sab  = (bf16_t*)alloc((size_t)MM * CS * 2);
  float*  b1f  = (float*) alloc((size_t)MM * 2 * CS * 4);
  float*  b2f  = (float*) alloc((size_t)MM * 2 * CS * 4);
  bf16_t* bmb  = (bf16_t*)alloc((size_t)MM * 2 * CS * 2);
  float*  tmf  = (float*) alloc((size_t)MM * CS * 4);

  auto grid1 = [](int n) { return dim3((n + 255) / 256); };
  auto tr = [&](const float* W, bf16_t* Wt, int K, int N) {
    k_transpose_bf16<<<grid1(K * N), 256, 0, stream>>>(W, Wt, K, N);
  };
  auto gemm = [&](const bf16_t* A, const bf16_t* Bt, const float* bias, float* C,
                  int M, int N, int K, int lda, int ldb, int ldc) {
    int waves = (M / 16) * (N / 16);
    k_gemm_bf16<<<dim3((waves + 7) / 8), 256, 0, stream>>>(A, Bt, bias, C, M, N, K, lda, ldb, ldc);
  };

  // weights -> bf16 transposed
  tr(Wq,  WqT,  CS, CS);      tr(Wkv, WkvT, CS, 2 * CS);
  tr(Wg,  WgT,  CS, CS);      tr(Wo,  WoT,  CS, CS);
  tr(Wc1, Wc1T, CCOND, CS);   tr(Wcn, WcnT, CCOND, CS);
  tr(Wc2, Wc2T, CCOND, CS);   tr(W1,  W1T,  CS, 2 * CS);
  tr(W2,  W2T,  CS, 2 * CS);  tr(Wb,  WbT,  2 * CS, CS);

  // x = LN(rigids_embed)
  k_layernorm<<<dim3(MM / 8), 256, 0, stream>>>(re, ln_g, ln_b, xn, MM, CS);

  // q, kv, gate projections (WMMA)
  gemm(xn, WqT,  nullptr, qf,  MM, CS,     CS, CS, CS, CS);
  gemm(xn, WkvT, nullptr, kvf, MM, 2 * CS, CS, CS, CS, 2 * CS);
  gemm(xn, WgT,  nullptr, gpf, MM, CS,     CS, CS, CS, CS);

  k_prep_attn<<<grid1(MM * CS), 256, 0, stream>>>(qf, kvf, qb, kb, vt);

  // pair bias + mask pre-fill: single WMMA-based pass over the 512MB framepair tensor
  k_pbias<<<dim3(BB * NN * NN / 128), 256, 0, stream>>>(fp, lnz_g, lnz_b, Wbias, mask, att);

  // logits += qk^T/sqrt(DH)
  {
    int waves = BB * HH * 64 * 64;
    k_attn_logits<<<dim3(waves / 8), 256, 0, stream>>>(qb, kb, att);
  }
  // softmax
  k_softmax<<<dim3(BB * HH * NN), 256, 0, stream>>>(att, attb);

  // o = A @ V per (b,h)  (WMMA, K=1024)
  for (int bh = 0; bh < BB * HH; ++bh) {
    int b = bh >> 3, h = bh & 7;
    gemm(attb + (size_t)bh * NN * NN, vt + (size_t)bh * DHD * NN, nullptr,
         of + (size_t)b * NN * CS + h * DHD, NN, DHD, NN, NN, NN, CS);
  }

  // gated output projection + residual
  k_ew_gate<<<grid1(MM * CS), 256, 0, stream>>>(of, gpf, ub, MM * CS);
  gemm(ub, WoT, nullptr, updf, MM, CS, CS, CS, CS, CS);
  k_ew_rig<<<grid1(MM * CS), 256, 0, stream>>>(re, updf, mask, rig, MM * CS);

  // conditioning
  k_layernorm<<<dim3(MM / 8), 256, 0, stream>>>(rig, nullptr, nullptr, snb, MM, CS);
  k_convert_bf16<<<grid1(BB * NRES * CCOND), 256, 0, stream>>>(s_in, sb, BB * NRES * CCOND);
  k_layernorm<<<dim3(BB * NRES / 8), 256, 0, stream>>>(s_in, lncg, nullptr, cnb, BB * NRES, CCOND);
  gemm(cnb, Wc1T, bc1,     c1f,  BB * NRES, CS, CCOND, CCOND, CCOND, CS);
  gemm(cnb, WcnT, nullptr, cnf,  BB * NRES, CS, CCOND, CCOND, CCOND, CS);
  gemm(sb,  Wc2T, bc2,     cgpf, BB * NRES, CS, CCOND, CCOND, CCOND, CS);
  k_ew_sa<<<grid1(MM * CS), 256, 0, stream>>>(snb, c1f, cnf, idx, mask, sab, MM * CS);

  // transition MLP
  gemm(sab, W1T, nullptr, b1f, MM, 2 * CS, CS, CS, CS, 2 * CS);
  gemm(sab, W2T, nullptr, b2f, MM, 2 * CS, CS, CS, CS, 2 * CS);
  k_ew_bmid<<<grid1(MM * 2 * CS), 256, 0, stream>>>(b1f, b2f, bmb, MM * 2 * CS);
  gemm(bmb, WbT, nullptr, tmf, MM, CS, 2 * CS, 2 * CS, 2 * CS, CS);

  k_ew_final<<<grid1(MM * CS), 256, 0, stream>>>(rig, tmf, cgpf, idx, mask, out, MM * CS);
}
